// FNO2D_33715493274138
// MI455X (gfx1250) — matrix operand order repaired and checked by reading, not compiled, not run on hardware
//
#include <hip/hip_runtime.h>
#include <math.h>

typedef __attribute__((ext_vector_type(16))) _Float16 v16h;
typedef __attribute__((ext_vector_type(8)))  _Float16 v8h;
typedef __attribute__((ext_vector_type(8)))  float    v8f;

#define NB 16
#define NH 256
#define NW 256
#define CH 64
#define NL 4
#define HW (NH*NW)
#define PI2_256 0.024543692606f   // 2*pi/256

__device__ __forceinline__ v8f wmma_f16(v16h a, v16h b, v8f c) {
  return __builtin_amdgcn_wmma_f32_16x16x32_f16(false, a, false, b, (short)0, c, false, false);
}

// A: 16x32 (MxK) row-major, K contiguous. Two aligned 16B vector loads per lane.
// lane<16: row=lane, halves = K{0..7,16..23}; lane>=16: row=lane-16, K{8..15,24..31}
// Requires: ld multiple of 8 halves, src 16B aligned.
__device__ __forceinline__ v16h load_A16(const _Float16* __restrict__ src, int ld) {
  int lane = threadIdx.x & 31;
  const _Float16* p = src + (lane & 15) * ld + ((lane < 16) ? 0 : 8);
  v8h lo = *(const v8h*)p;
  v8h hi = *(const v8h*)(p + 16);
  return __builtin_shufflevector(lo, hi, 0,1,2,3,4,5,6,7,8,9,10,11,12,13,14,15);
}

// B: 32x16 (KxN) stored TRANSPOSED as src[col*ld + k] (k contiguous).
// lane<16: col=lane, K{0..7,16..23}; lane>=16: col=lane-16, K{8..15,24..31}. Two aligned 16B loads.
// Requires: ld multiple of 8 halves, src 16B aligned.
__device__ __forceinline__ v16h load_Bt(const _Float16* __restrict__ src, int ld) {
  int lane = threadIdx.x & 31;
  const _Float16* p = src + (lane & 15) * ld + ((lane < 16) ? 0 : 8);
  v8h lo = *(const v8h*)p;
  v8h hi = *(const v8h*)(p + 16);
  return __builtin_shufflevector(lo, hi, 0,1,2,3,4,5,6,7,8,9,10,11,12,13,14,15);
}

// D: 16x16 f32. lane<16: N=lane, M=vgpr; lane>=16: N=lane-16, M=8+vgpr
__device__ __forceinline__ void store_D(float* __restrict__ dst, int ld, v8f d) {
  int lane = threadIdx.x & 31;
  int col = lane & 15;
  int rb = (lane < 16) ? 0 : 8;
#pragma unroll
  for (int i = 0; i < 8; ++i) dst[(rb + i) * ld + col] = d[i];
}

__device__ __forceinline__ float gelu(float v) {
  return 0.5f * v * (1.0f + erff(v * 0.70710678118f));
}

// ---------------- tables (f16), exact mod-256 phase reduction ----------------
// FXp [2 nt][8 kc][16 col][32 k]: forward x-DFT B, fragment-packed. nt0=cos, nt1=-sin
// AY  [2][32][512] row-major (A operand): part0 -> ft_re, part1 -> ft_im
// EU  [2][256][64] row-major (A operand): part0 -> u_re, part1 -> u_im
// T2p [16 nt][16 col][32 k]: inverse x B, fragment-packed (K rows 0..15 cos, 16..31 -sin)
__global__ void k_tables(_Float16* FXp, _Float16* AY, _Float16* EU, _Float16* T2p) {
  for (int n = blockIdx.x * blockDim.x + threadIdx.x; n < 81920; n += gridDim.x * blockDim.x) {
    if (n < 8192) {                       // FX packed
      int kk = n & 31, col = (n >> 5) & 15, kc = (n >> 9) & 7, w = n >> 12;
      int x = kc * 32 + kk;
      float th = (float)((col * x) & 255) * PI2_256;
      FXp[n] = (_Float16)((w == 0) ? cosf(th) : -sinf(th));
    } else if (n < 40960) {               // AY row-major
      int n2 = n - 8192;
      int part = n2 >> 14, r = (n2 >> 9) & 31, col = n2 & 511;
      int yy = col & 255, h2 = col >> 8;
      int ky = (r < 16) ? r : (224 + r);
      float th = (float)((ky * yy) & 255) * PI2_256;
      float v = (part == 0) ? (h2 ? sinf(th) : cosf(th))
                            : (h2 ? cosf(th) : -sinf(th));
      AY[n2] = (_Float16)v;
    } else if (n < 73728) {               // EU row-major
      int n3 = n - 40960;
      int part = n3 >> 14, y = (n3 >> 6) & 255, m6 = n3 & 63;
      int m = m6 & 31, ish = m6 >> 5;
      int ky = (m < 16) ? m : (224 + m);
      float th = (float)((ky * y) & 255) * PI2_256;
      float v = (part == 0) ? (ish ? -sinf(th) : cosf(th))
                            : (ish ? cosf(th)  : sinf(th));
      EU[n3] = (_Float16)v;
    } else {                              // T2 packed
      int n4 = n - 73728;
      int kk = n4 & 31, col = (n4 >> 5) & 15, nt = n4 >> 9;
      int x = nt * 16 + col;
      int k = kk & 15;
      float cf = ((k == 0) ? 1.0f : 2.0f) / (float)HW;
      float th = (float)((k * x) & 255) * PI2_256;
      T2p[n4] = (_Float16)((kk < 16) ? (cf * cosf(th)) : (-cf * sinf(th)));
    }
  }
}

// spectral weights -> f16, fragment-packed per mode: [L][blk][ky][kx][w2][kc2][dup2][col16][k32] (bit 9 ignored: o<32 only)
__global__ void k_prepw(const float* __restrict__ w1r, const float* __restrict__ w1i,
                        const float* __restrict__ w2r, const float* __restrict__ w2i,
                        _Float16* __restrict__ whre, _Float16* __restrict__ whim) {
  int idx = blockIdx.x * blockDim.x + threadIdx.x;   // 8,388,608 total
  int m12 = idx & 4095;
  int kk = m12 & 31, col = (m12 >> 5) & 15, kc = (m12 >> 10) & 1, w = m12 >> 11;
  int i = kc * 32 + kk;                 // K index (input channel)
  int o = w * 16 + col;                 // N index (output channel)
  int kx = (idx >> 12) & 15, kyl = (idx >> 16) & 15;
  int blkb = (idx >> 20) & 1, l = idx >> 21;
  int s = (((l * 64 + i) * 64 + o) * 16 + kyl) * 16 + kx;
  whre[idx] = (_Float16)(blkb ? w2r[s] : w1r[s]);
  whim[idx] = (_Float16)(blkb ? w2i[s] : w1i[s]);
}

// cw -> f16 row-major [L][o][i] (A operand); p1 -> fragment-packed [nt4][kc2][dup2][col16][k32] (bit 9 ignored: j<64 only)
__global__ void k_preps(const float* __restrict__ cw, const float* __restrict__ p1w,
                        _Float16* __restrict__ cwh, _Float16* __restrict__ p1hp) {
  int idx = blockIdx.x * blockDim.x + threadIdx.x;
  if (idx < 16384) {
    cwh[idx] = (_Float16)cw[idx];
  } else if (idx < 24576) {
    int p = idx - 16384;
    int kk = p & 31, col = (p >> 5) & 15, kc = (p >> 10) & 1, nt = p >> 11;
    int c = kc * 32 + kk;
    int j = nt * 16 + col;
    p1hp[p] = (_Float16)p1w[c * 128 + j];
  }
}

// lift: h[b,c,y,x] = x[b,y,x,0]*lw[0,c] + x[b,y,x,1]*lw[1,c] + lb[c]   (h stored f16)
__global__ void k_lift(const float* __restrict__ xin, const float* __restrict__ lw,
                       const float* __restrict__ lb, _Float16* __restrict__ h) {
  int idx = blockIdx.x * blockDim.x + threadIdx.x;   // 67,108,864 total
  int x = idx & 255, y = (idx >> 8) & 255, c = (idx >> 16) & 63, b = idx >> 22;
  int pos = (b << 16) | (y << 8) | x;
  h[idx] = (_Float16)(xin[pos * 2] * lw[c] + xin[pos * 2 + 1] * lw[64 + c] + lb[c]);
}

// stage 1 fwd: tmp[row(262144)][32] = Hrows[row][256] @ FX[256][32]
// h is f16: A fragments load straight from global, no LDS. 1 wave, A reused for both N-tiles.
__global__ void k_dftx(const _Float16* __restrict__ h, const _Float16* __restrict__ FXp,
                       float* __restrict__ tmp) {
  int blk = blockIdx.x;             // 16384 M-tiles
  const _Float16* hg = h + (long long)blk * 4096;
  v8f acc0 = {}, acc1 = {};
#pragma unroll
  for (int kc = 0; kc < 8; ++kc) {
    v16h a = load_A16(hg + kc * 32, 256);
    acc0 = wmma_f16(a, load_Bt(FXp + kc * 512, 32), acc0);         // re cols
    acc1 = wmma_f16(a, load_Bt(FXp + (8 + kc) * 512, 32), acc1);   // im cols
  }
  store_D(tmp + (long long)blk * 512, 32, acc0);
  store_D(tmp + (long long)blk * 512 + 16, 32, acc1);
}

// stage 2 fwd per (b,c): ft[64][16] (rows 0..31 re, 32..63 im) = AY[part][32][512] @ stack(tmp)
__global__ void k_dfty(const float* __restrict__ tmp, const _Float16* __restrict__ AY,
                       float* __restrict__ ft) {
  __shared__ __align__(16) _Float16 Bt[16][520];
  int blk = blockIdx.x;   // 1024 = b*CH + c
  int tid = threadIdx.x;  // 64
  const float4* s4 = (const float4*)(tmp + blk * 8192);
  for (int t = tid; t < 2048; t += 64) {
    float4 v = s4[t];
    int yy = t >> 3, c0 = (t & 7) * 4;
    float vv[4] = {v.x, v.y, v.z, v.w};
#pragma unroll
    for (int j = 0; j < 4; ++j) {
      int c = c0 + j;
      if (c < 16) Bt[c][yy] = (_Float16)vv[j];
      else        Bt[c - 16][256 + yy] = (_Float16)vv[j];
    }
  }
  __syncthreads();
  int w = tid >> 5;       // part 0=re 1=im
  const _Float16* ayp = AY + w * 32 * 512;
  for (int mt = 0; mt < 2; ++mt) {
    v8f acc = {};
#pragma unroll
    for (int kc = 0; kc < 16; ++kc) {
      v16h a = load_A16(ayp + (mt * 16) * 512 + kc * 32, 512);
      v16h b = load_Bt(&Bt[0][kc * 32], 520);
      acc = wmma_f16(a, b, acc);
    }
    store_D(ft + blk * 1024 + (w * 32 + mt * 16) * 16, 16, acc);
  }
}

// spectral: per mode (ky,kx): out[b,o] = ft[b,i] * W[i,o]  (complex, batched over b=16)
__global__ void k_spec(const float* __restrict__ ft, const _Float16* __restrict__ whre,
                       const _Float16* __restrict__ whim, float* __restrict__ oft, int l) {
  __shared__ __align__(16) _Float16 Asr[16][72];
  __shared__ __align__(16) _Float16 Asi[16][72];
  int blk = blockIdx.x;   // 512 modes
  int ky = blk >> 4, kx = blk & 15;
  int tid = threadIdx.x;  // 128 (4 waves)
  for (int t = tid; t < 1024; t += 128) {
    int bb = t >> 6, i = t & 63;
    int base = ((bb * CH + i) * 64 + ky) * 16 + kx;
    Asr[bb][i] = (_Float16)ft[base];
    Asi[bb][i] = (_Float16)ft[base + 512];
  }
  __syncthreads();
  int mode = ((l * 2 + (ky >> 4)) * 16 + (ky & 15)) * 16 + kx;
  const _Float16* wre = whre + (long long)mode * 4096;
  const _Float16* wim = whim + (long long)mode * 4096;
  int w = tid >> 5;       // o-tile
  v8f accRR = {}, accII = {}, accIM = {};
#pragma unroll
  for (int kc = 0; kc < 2; ++kc) {
    v16h ar = load_A16(&Asr[0][kc * 32], 72);
    v16h ai = load_A16(&Asi[0][kc * 32], 72);
    v16h br = load_Bt(wre + (w * 2 + kc) * 512, 32);
    v16h bi = load_Bt(wim + (w * 2 + kc) * 512, 32);
    accRR = wmma_f16(ar, br, accRR);
    accII = wmma_f16(ai, bi, accII);
    accIM = wmma_f16(ar, bi, accIM);
    accIM = wmma_f16(ai, br, accIM);
  }
  int lane = tid & 31;
  int col = lane & 15, rb = (lane < 16) ? 0 : 8;
  int o = w * 16 + col;
#pragma unroll
  for (int i = 0; i < 8; ++i) {
    int bb = rb + i;
    int base = ((bb * CH + o) * 64 + ky) * 16 + kx;
    oft[base] = accRR[i] - accII[i];
    oft[base + 512] = accIM[i];
  }
}

// inverse y per (b,c): U[256][32] (cols 0..15 u_re, 16..31 u_im) = EU[part][256][64] @ oft-stack
__global__ void k_idy(const float* __restrict__ oft, const _Float16* __restrict__ EU,
                      float* __restrict__ U) {
  __shared__ __align__(16) _Float16 Bt[16][72];
  int blk = blockIdx.x;   // 1024
  int tid = threadIdx.x;  // 64
  const float4* s4 = (const float4*)(oft + blk * 1024);
  for (int t = tid; t < 256; t += 64) {
    float4 v = s4[t];
    int e = t * 4;
    int k = e >> 4, col = e & 15;
    Bt[col][k] = (_Float16)v.x; Bt[col + 1][k] = (_Float16)v.y;
    Bt[col + 2][k] = (_Float16)v.z; Bt[col + 3][k] = (_Float16)v.w;
  }
  __syncthreads();
  int w = tid >> 5;
  const _Float16* eup = EU + w * 256 * 64;
  v16h b0 = load_Bt(&Bt[0][0], 72);
  v16h b1 = load_Bt(&Bt[0][32], 72);
  for (int mt = 0; mt < 16; ++mt) {
    v8f acc = {};
    v16h a0 = load_A16(eup + (mt * 16) * 64, 64);
    acc = wmma_f16(a0, b0, acc);
    v16h a1 = load_A16(eup + (mt * 16) * 64 + 32, 64);
    acc = wmma_f16(a1, b1, acc);
    store_D(U + blk * 8192 + (mt * 16) * 32 + w * 16, 32, acc);
  }
}

// fused inverse-x + pointwise conv + gelu:
//   h_next[o][y][x] = gelu( cw@h + cb + U[o][y][:] @ T2[:, x] )
// per workgroup: (b, 16y, 16x) tile, all 64 o streamed in groups of 16.
// x1 accumulates into the SAME wmma accumulator as x2 (per-y tiles, M=o, N=x).
__global__ void k_comb(const _Float16* __restrict__ h, const float* __restrict__ U,
                       const _Float16* __restrict__ cwh, const float* __restrict__ cb,
                       const _Float16* __restrict__ T2p, _Float16* __restrict__ hnext, int l) {
  __shared__ __align__(16) _Float16 hst[256][72];     // B-layout [pix][i], +8 pad (36.9KB)
  __shared__ __align__(16) _Float16 Us[16][16][40];   // [y][o][k], +8 pad (20.5KB)
  int blk = blockIdx.x;          // 4096 = b(16) * yt(16) * xt(16)
  int b = blk >> 8, yt = (blk >> 4) & 15, xt = blk & 15;
  int tid = threadIdx.x;         // 256, 8 waves
  // stage h patch [64 i][16y x 16x] -> hst[pix][i]  (f16 source: raw 16B copies)
  const _Float16* hb = h + (long long)b * CH * HW + (yt * 16) * NW + xt * 16;
  for (int t = tid; t < 2048; t += 256) {
    int i = t >> 5, rem = t & 31, y = rem >> 1, x8 = (rem & 1) * 8;
    v8h v = *(const v8h*)(hb + (long long)i * HW + y * NW + x8);
    int pix = y * 16 + x8;
#pragma unroll
    for (int j = 0; j < 8; ++j) hst[pix + j][i] = v[j];
  }
  int w = tid >> 5;
  int lane = tid & 31;
  int xx = lane & 15, rb = (lane < 16) ? 0 : 8;
  v16h b_t2 = load_Bt(T2p + xt * 512, 32);   // same inverse-x B fragment for every tile
  const _Float16* cwb = cwh + l * 4096;
  const float* cbl = cb + l * 64;
  for (int og = 0; og < 4; ++og) {
    __syncthreads();   // hst ready (og=0) / previous og done with Us
    const float* ub = U + ((long long)(b * CH + og * 16) * NH + yt * 16) * 32;
    for (int t = tid; t < 2048; t += 256) {
      int o = t >> 7, rem = t & 127, y = rem >> 3, k4 = (rem & 7) * 4;
      float4 v = *(const float4*)(ub + ((long long)o * NH + y) * 32 + k4);
      Us[y][o][k4]     = (_Float16)v.x; Us[y][o][k4 + 1] = (_Float16)v.y;
      Us[y][o][k4 + 2] = (_Float16)v.z; Us[y][o][k4 + 3] = (_Float16)v.w;
    }
    __syncthreads();
    v16h a0 = load_A16(cwb + (og * 16) * 64, 64);
    v16h a1 = load_A16(cwb + (og * 16) * 64 + 32, 64);
#pragma unroll
    for (int j = 0; j < 2; ++j) {
      int y = w * 2 + j;
      v8f acc = {};
      acc = wmma_f16(a0, load_Bt(&hst[y * 16][0], 72), acc);   // x2: cw @ h (K=0..31)
      acc = wmma_f16(a1, load_Bt(&hst[y * 16][32], 72), acc);  // x2: cw @ h (K=32..63)
      acc = wmma_f16(load_A16(&Us[y][0][0], 40), b_t2, acc);   // x1: U @ T2, same tile
      long long obase = ((long long)(b * CH + og * 16) * NH + yt * 16 + y) * NW + xt * 16 + xx;
#pragma unroll
      for (int i = 0; i < 8; ++i) {
        int o = rb + i;
        hnext[obase + (long long)o * HW] = (_Float16)gelu(acc[i] + cbl[og * 16 + o]);
      }
    }
  }
}

// projection: out = gelu(h^T @ p1 + b1) @ p2 + b2, fused per 16-pixel tile
__global__ void k_proj(const _Float16* __restrict__ h, const _Float16* __restrict__ p1hp,
                       const float* __restrict__ p1b, const float* __restrict__ p2w,
                       const float* __restrict__ p2b, float* __restrict__ out) {
  __shared__ __align__(16) _Float16 hsa[16][72];
  __shared__ float ts[16][128];
  int blk = blockIdx.x;       // 65536 pixel tiles
  int bb = blk >> 12;
  long long rem0 = (long long)(blk & 4095) * 16;
  int tid = threadIdx.x;      // 128
  const _Float16* src = h + (long long)bb * CH * HW + rem0;
  for (int t = tid; t < 128; t += 128) {
    int c = t >> 1, p8 = (t & 1) * 8;
    v8h v = *(const v8h*)(src + (long long)c * HW + p8);
#pragma unroll
    for (int j = 0; j < 8; ++j) hsa[p8 + j][c] = v[j];
  }
  __syncthreads();
  int w = tid >> 5;
  int lane = tid & 31;
  int col = lane & 15, rb = (lane < 16) ? 0 : 8;
  for (int nt = 2 * w; nt < 2 * w + 2; ++nt) {
    v8f acc = {};
#pragma unroll
    for (int kc = 0; kc < 2; ++kc) {
      v16h a = load_A16(&hsa[0][kc * 32], 72);
      v16h bf = load_Bt(p1hp + (nt * 2 + kc) * 512, 32);
      acc = wmma_f16(a, bf, acc);
    }
    int j = nt * 16 + col;
#pragma unroll
    for (int i = 0; i < 8; ++i) ts[rb + i][j] = gelu(acc[i] + p1b[j]);
  }
  __syncthreads();
  if (tid < 32) {
    int pix = tid >> 1, co = tid & 1;
    float s = p2b[co];
    for (int j = 0; j < 128; ++j) s += ts[pix][j] * p2w[j * 2 + co];
    out[((long long)blk * 16 + pix) * 2 + co] = s;
  }
}

extern "C" void kernel_launch(void* const* d_in, const int* in_sizes, int n_in,
                              void* d_out, int out_size, void* d_ws, size_t ws_size,
                              hipStream_t stream) {
  (void)in_sizes; (void)n_in; (void)out_size; (void)ws_size;
  const float* xin = (const float*)d_in[0];
  const float* lw  = (const float*)d_in[1];
  const float* lb  = (const float*)d_in[2];
  const float* w1r = (const float*)d_in[3];
  const float* w1i = (const float*)d_in[4];
  const float* w2r = (const float*)d_in[5];
  const float* w2i = (const float*)d_in[6];
  const float* cw  = (const float*)d_in[7];
  const float* cb  = (const float*)d_in[8];
  const float* p1w = (const float*)d_in[9];
  const float* p1b = (const float*)d_in[10];
  const float* p2w = (const float*)d_in[11];
  const float* p2b = (const float*)d_in[12];
  float* out = (float*)d_out;

  char* ws = (char*)d_ws;
  size_t off = 0;
  auto alloc = [&](size_t bytes) { char* p = ws + off; off += (bytes + 255) & ~(size_t)255; return p; };
  _Float16* hA   = (_Float16*)alloc(134217728);  // [B][64][256][256] f16
  _Float16* hB   = (_Float16*)alloc(134217728);
  float*    tmpU = (float*)alloc(33554432);      // reused: fwd tmp / inverse U (L2-resident)
  float*    ft   = (float*)alloc(4194304);       // [B][64][64][16]
  float*    oft  = (float*)alloc(4194304);
  _Float16* FXp  = (_Float16*)alloc(16384);
  _Float16* AY   = (_Float16*)alloc(65536);
  _Float16* EU   = (_Float16*)alloc(65536);
  _Float16* T2p  = (_Float16*)alloc(16384);
  _Float16* whre = (_Float16*)alloc(16777216);
  _Float16* whim = (_Float16*)alloc(16777216);
  _Float16* cwh  = (_Float16*)alloc(32768);
  _Float16* p1hp = (_Float16*)alloc(16384);

  k_tables<<<320, 256, 0, stream>>>(FXp, AY, EU, T2p);
  k_prepw<<<32768, 256, 0, stream>>>(w1r, w1i, w2r, w2i, whre, whim);
  k_preps<<<96, 256, 0, stream>>>(cw, p1w, cwh, p1hp);
  k_lift<<<262144, 256, 0, stream>>>(xin, lw, lb, hA);

  _Float16* hcur = hA;
  _Float16* hoth = hB;
  for (int l = 0; l < NL; ++l) {
    k_dftx<<<16384, 32, 0, stream>>>(hcur, FXp, tmpU);
    k_dfty<<<1024, 64, 0, stream>>>(tmpU, AY, ft);
    k_spec<<<512, 128, 0, stream>>>(ft, whre, whim, oft, l);
    k_idy<<<1024, 64, 0, stream>>>(oft, EU, tmpU);
    k_comb<<<4096, 256, 0, stream>>>(hcur, tmpU, cwh, cb, T2p, hoth, l);
    _Float16* t = hcur; hcur = hoth; hoth = t;
  }
  k_proj<<<65536, 128, 0, stream>>>(hcur, p1hp, p1b, p2w, p2b, out);
}
